// Global_SlidingWindow_Attention_5042291605860
// MI455X (gfx1250) — matrix operand order repaired and checked by reading, not compiled, not run on hardware
//
#include <hip/hip_runtime.h>
#include <hip/hip_bf16.h>

typedef __attribute__((ext_vector_type(16))) _Float16 v16h;
typedef __attribute__((ext_vector_type(8)))  _Float16 v8h;
typedef __attribute__((ext_vector_type(8)))  float    v8f;

#define HEADS 16
#define HD    64
#define SEQ   2048
#define BATCH 2
#define HID   1024

union H16 { v16h v; v8h h[2]; };

// async global->LDS copy, 16B per lane, tracked by ASYNCcnt (CDNA5)
#define ASYNC_LD_B128(ldsoff, gptr)                                         \
    asm volatile("global_load_async_to_lds_b128 %0, %1, off"                \
                 :: "v"(ldsoff), "v"(gptr) : "memory")
#define WAIT_ASYNC(n)                                                       \
    asm volatile("s_wait_asynccnt " #n ::: "memory")

// ---------------------------------------------------------------------------
// fp32 -> f16 conversion
// ---------------------------------------------------------------------------
__global__ __launch_bounds__(256) void cvt_f32_f16(const float* __restrict__ in,
                                                   _Float16* __restrict__ out, int n) {
    int i = blockIdx.x * 256 + threadIdx.x;
    if (i < n) out[i] = (_Float16)in[i];
}

// ---------------------------------------------------------------------------
// GEMM: D[M,N] = A[M,K] @ W[N,K]^T + bias[N]
// Block = 8 waves = 128(M) x 64(N).  The 64x32 weight slab per K-step is
// shared by all 8 waves -> stage it into LDS with double-buffered
// global_load_async_to_lds_b128 (one 16B load per thread per step).
// MODE 0: f16 row-major out.  MODE 1: f16 transposed out to Vt[b][h][d][s].
// MODE 2: fp32 row-major out.
// ---------------------------------------------------------------------------
template <int MODE>
__global__ __launch_bounds__(256) void gemm_wmma(const _Float16* __restrict__ A,
                                                 const _Float16* __restrict__ W,
                                                 const float* __restrict__ bias,
                                                 void* __restrict__ outp,
                                                 int M, int N, int K) {
    __shared__ _Float16 wtile[2][64][40];   // 32 used + 8 pad halfs per row

    const int tid  = threadIdx.x;
    const int lane = tid & 31;
    const int w    = tid >> 5;
    const int g    = lane >> 4;     // K-half selector
    const int c    = lane & 15;     // row (A) / col (B,C)

    const int m0 = (blockIdx.x * 8 + w) * 16;
    const int n0 = blockIdx.y * 64;

    // cooperative async-copy assignment: 64 rows x 64B, 4 threads/row x 16B
    const int lrow = tid >> 2;
    const int lcol = (tid & 3) * 8;
    const _Float16* wsrc = W + (size_t)(n0 + lrow) * K + lcol;
    const unsigned  ldst0 = (unsigned)(size_t)&wtile[0][lrow][lcol];
    const unsigned  ldst1 = (unsigned)(size_t)&wtile[1][lrow][lcol];

    ASYNC_LD_B128(ldst0, wsrc);               // prime slab k=0

    v8f acc[4] = {v8f{}, v8f{}, v8f{}, v8f{}};
    const _Float16* arow = A + (size_t)(m0 + c) * K;

    for (int k = 0; k < K; k += 32) {
        const int cur = (k >> 5) & 1;
        if (k + 32 < K) {
            ASYNC_LD_B128(cur ? ldst0 : ldst1, wsrc + k + 32);
            WAIT_ASYNC(0x1);                  // current slab landed (in-order)
        } else {
            WAIT_ASYNC(0x0);
        }
        __syncthreads();

        H16 af;
        af.h[0] = *(const v8h*)(arow + k + 8 * g);
        af.h[1] = *(const v8h*)(arow + k + 16 + 8 * g);
#pragma unroll
        for (int nt = 0; nt < 4; ++nt) {
            H16 bf;
            bf.h[0] = *(const v8h*)(&wtile[cur][nt * 16 + c][8 * g]);
            bf.h[1] = *(const v8h*)(&wtile[cur][nt * 16 + c][16 + 8 * g]);
            acc[nt] = __builtin_amdgcn_wmma_f32_16x16x32_f16(
                false, af.v, false, bf.v, (short)0, acc[nt], false, false);
        }
        __syncthreads();                      // protect cur^1 before overwrite
    }

    if (MODE == 0) {
        _Float16* out = (_Float16*)outp;
#pragma unroll
        for (int nt = 0; nt < 4; ++nt) {
            int col = n0 + nt * 16 + c;
            float bv = bias[col];
#pragma unroll
            for (int r = 0; r < 8; ++r) {
                int row = m0 + r + 8 * g;
                out[(size_t)row * N + col] = (_Float16)(acc[nt][r] + bv);
            }
        }
    } else if (MODE == 1) {
        // Vt layout: [b][h][d][s], s contiguous.  tokens m0..m0+15 share b.
        _Float16* vt = (_Float16*)outp;
        int b  = m0 / SEQ;
        int s0 = m0 % SEQ;
#pragma unroll
        for (int nt = 0; nt < 4; ++nt) {
            int d  = n0 + nt * 16 + c;       // global hidden column
            int h  = d >> 6;
            int dd = d & 63;
            float bv = bias[d];
            v8h pack;
#pragma unroll
            for (int r = 0; r < 8; ++r) pack[r] = (_Float16)(acc[nt][r] + bv);
            _Float16* dst = vt + ((size_t)((b * HEADS + h) * HD + dd)) * SEQ + s0 + 8 * g;
            *(v8h*)dst = pack;
        }
    } else {
        float* out = (float*)outp;
#pragma unroll
        for (int nt = 0; nt < 4; ++nt) {
            int col = n0 + nt * 16 + c;
            float bv = bias[col];
#pragma unroll
            for (int r = 0; r < 8; ++r) {
                int row = m0 + r + 8 * g;
                out[(size_t)row * N + col] = acc[nt][r] + bv;
            }
        }
    }
}

// ---------------------------------------------------------------------------
// Flash-style masked attention.  Mask (S=2048): |i-j|<=64  OR  i%64==32.
// One wave per 16-query tile; online softmax; 32-key chunks (8 wmma each).
// ---------------------------------------------------------------------------
__global__ __launch_bounds__(256) void attn_wmma(const _Float16* __restrict__ qh,
                                                 const _Float16* __restrict__ kh,
                                                 const _Float16* __restrict__ vt,
                                                 _Float16* __restrict__ ctx) {
    __shared__ _Float16 plds[8][16][40];   // padded to dodge bank conflicts

    const int lane = threadIdx.x & 31;
    const int w    = threadIdx.x >> 5;
    const int g    = lane >> 4;
    const int c    = lane & 15;

    const int bh = blockIdx.y;
    const int b  = bh >> 4;
    const int h  = bh & 15;
    const int qm = (blockIdx.x * 8 + w) * 16;

    // Q as A-fragments (two K=32 slabs of head_dim=64)
    const _Float16* qbase = qh + (size_t)(b * SEQ + qm + c) * HID + h * HD;
    H16 qa0, qa1;
    qa0.h[0] = *(const v8h*)(qbase + 16 * g);
    qa0.h[1] = *(const v8h*)(qbase + 16 * g + 8);
    qa1.h[0] = *(const v8h*)(qbase + 32 + 16 * g);
    qa1.h[1] = *(const v8h*)(qbase + 32 + 16 * g + 8);

    v8f o0 = {}, o1 = {}, o2 = {}, o3 = {};
    float mrow[8], lrow[8];
#pragma unroll
    for (int r = 0; r < 8; ++r) { mrow[r] = -1e30f; lrow[r] = 0.0f; }

    const bool globalRow = (qm & 63) == 32;   // tile contains a global token row
    const _Float16* kb0 = kh + (size_t)(b * SEQ) * HID + h * HD;
    const _Float16* vtb = vt + (size_t)(b * HEADS + h) * HD * SEQ;

    for (int kc = 0; kc < SEQ; kc += 32) {
        if (!globalRow && !(kc + 31 >= qm - 64 && kc <= qm + 79)) continue;

        // scores: two 16x16 tiles (keys kc..+15, kc+16..+31)
        v8f s0 = {}, s1 = {};
        {
            const _Float16* kr = kb0 + (size_t)(kc + c) * HID;
            v16h f0 = *(const v16h*)(kr + 16 * g);
            v16h f1 = *(const v16h*)(kr + 32 + 16 * g);
            s0 = __builtin_amdgcn_wmma_f32_16x16x32_f16(false, qa0.v, false, f0, (short)0, s0, false, false);
            s0 = __builtin_amdgcn_wmma_f32_16x16x32_f16(false, qa1.v, false, f1, (short)0, s0, false, false);
        }
        {
            const _Float16* kr = kb0 + (size_t)(kc + 16 + c) * HID;
            v16h f0 = *(const v16h*)(kr + 16 * g);
            v16h f1 = *(const v16h*)(kr + 32 + 16 * g);
            s1 = __builtin_amdgcn_wmma_f32_16x16x32_f16(false, qa0.v, false, f0, (short)0, s1, false, false);
            s1 = __builtin_amdgcn_wmma_f32_16x16x32_f16(false, qa1.v, false, f1, (short)0, s1, false, false);
        }

        // mask + online softmax (row stats per half-wave; N-dim reduce via shfl)
#pragma unroll
        for (int r = 0; r < 8; ++r) {
            int i = qm + r + 8 * g;
            bool gi = (i & 63) == 32;
            float v0 = s0[r] * 0.125f;
            float v1 = s1[r] * 0.125f;
            int d0 = i - (kc + c);      if (d0 < 0) d0 = -d0;
            int d1 = i - (kc + 16 + c); if (d1 < 0) d1 = -d1;
            if (!gi && d0 > 64) v0 = -1e9f;
            if (!gi && d1 > 64) v1 = -1e9f;

            float t = fmaxf(v0, v1);
            t = fmaxf(t, __shfl_xor(t, 1, 32));
            t = fmaxf(t, __shfl_xor(t, 2, 32));
            t = fmaxf(t, __shfl_xor(t, 4, 32));
            t = fmaxf(t, __shfl_xor(t, 8, 32));

            float mn    = fmaxf(mrow[r], t);
            float alpha = __expf(mrow[r] - mn);
            mrow[r] = mn;

            float e0 = __expf(v0 - mn);
            float e1 = __expf(v1 - mn);
            float rs = e0 + e1;
            rs += __shfl_xor(rs, 1, 32);
            rs += __shfl_xor(rs, 2, 32);
            rs += __shfl_xor(rs, 4, 32);
            rs += __shfl_xor(rs, 8, 32);
            lrow[r] = lrow[r] * alpha + rs;

            o0[r] *= alpha; o1[r] *= alpha; o2[r] *= alpha; o3[r] *= alpha;

            plds[w][r + 8 * g][c]      = (_Float16)e0;
            plds[w][r + 8 * g][16 + c] = (_Float16)e1;
        }

        // P (16x32) back as A-fragment from LDS
        H16 pa;
        pa.h[0] = *(const v8h*)(&plds[w][c][16 * g]);
        pa.h[1] = *(const v8h*)(&plds[w][c][16 * g + 8]);

        // O += P @ V   (B-fragments straight out of Vt rows, contiguous keys)
        {
            v16h vb = *(const v16h*)(vtb + (size_t)(c) * SEQ + kc + 16 * g);
            o0 = __builtin_amdgcn_wmma_f32_16x16x32_f16(false, pa.v, false, vb, (short)0, o0, false, false);
        }
        {
            v16h vb = *(const v16h*)(vtb + (size_t)(16 + c) * SEQ + kc + 16 * g);
            o1 = __builtin_amdgcn_wmma_f32_16x16x32_f16(false, pa.v, false, vb, (short)0, o1, false, false);
        }
        {
            v16h vb = *(const v16h*)(vtb + (size_t)(32 + c) * SEQ + kc + 16 * g);
            o2 = __builtin_amdgcn_wmma_f32_16x16x32_f16(false, pa.v, false, vb, (short)0, o2, false, false);
        }
        {
            v16h vb = *(const v16h*)(vtb + (size_t)(48 + c) * SEQ + kc + 16 * g);
            o3 = __builtin_amdgcn_wmma_f32_16x16x32_f16(false, pa.v, false, vb, (short)0, o3, false, false);
        }
    }

    // normalize and emit ctx (f16, [B*S][HID])
#pragma unroll
    for (int r = 0; r < 8; ++r) {
        float inv = 1.0f / lrow[r];
        _Float16* cb = ctx + (size_t)(b * SEQ + qm + r + 8 * g) * HID + h * HD;
        cb[c]      = (_Float16)(o0[r] * inv);
        cb[16 + c] = (_Float16)(o1[r] * inv);
        cb[32 + c] = (_Float16)(o2[r] * inv);
        cb[48 + c] = (_Float16)(o3[r] * inv);
    }
}

// ---------------------------------------------------------------------------
extern "C" void kernel_launch(void* const* d_in, const int* in_sizes, int n_in,
                              void* d_out, int out_size, void* d_ws, size_t ws_size,
                              hipStream_t stream) {
    const float* x  = (const float*)d_in[0];
    const float* Wq = (const float*)d_in[1];
    const float* bq = (const float*)d_in[2];
    const float* Wk = (const float*)d_in[3];
    const float* bk = (const float*)d_in[4];
    const float* Wv = (const float*)d_in[5];
    const float* bv = (const float*)d_in[6];
    const float* Wo = (const float*)d_in[7];
    const float* bo = (const float*)d_in[8];

    const int M = BATCH * SEQ;   // 4096
    const int N = HID;           // 1024
    const int K = HID;           // 1024

    char* ws = (char*)d_ws;
    size_t off = 0;
    _Float16* xh  = (_Float16*)(ws + off); off += (size_t)M * K * 2;       // 8 MB
    _Float16* wqh = (_Float16*)(ws + off); off += (size_t)N * K * 2;       // 2 MB
    _Float16* wkh = (_Float16*)(ws + off); off += (size_t)N * K * 2;
    _Float16* wvh = (_Float16*)(ws + off); off += (size_t)N * K * 2;
    _Float16* woh = (_Float16*)(ws + off); off += (size_t)N * K * 2;
    _Float16* qh  = (_Float16*)(ws + off); off += (size_t)M * N * 2;       // 8 MB
    _Float16* kh  = (_Float16*)(ws + off); off += (size_t)M * N * 2;
    _Float16* vt  = (_Float16*)(ws + off); off += (size_t)BATCH * HEADS * HD * SEQ * 2;
    _Float16* ctx = (_Float16*)(ws + off); off += (size_t)M * N * 2;

    cvt_f32_f16<<<(M * K + 255) / 256, 256, 0, stream>>>(x,  xh,  M * K);
    cvt_f32_f16<<<(N * K + 255) / 256, 256, 0, stream>>>(Wq, wqh, N * K);
    cvt_f32_f16<<<(N * K + 255) / 256, 256, 0, stream>>>(Wk, wkh, N * K);
    cvt_f32_f16<<<(N * K + 255) / 256, 256, 0, stream>>>(Wv, wvh, N * K);
    cvt_f32_f16<<<(N * K + 255) / 256, 256, 0, stream>>>(Wo, woh, N * K);

    dim3 gg(M / 128, N / 64);          // 32 x 16 blocks, 8 waves each
    gemm_wmma<0><<<gg, 256, 0, stream>>>(xh, wqh, bq, (void*)qh, M, N, K);
    gemm_wmma<0><<<gg, 256, 0, stream>>>(xh, wkh, bk, (void*)kh, M, N, K);
    gemm_wmma<1><<<gg, 256, 0, stream>>>(xh, wvh, bv, (void*)vt, M, N, K);

    attn_wmma<<<dim3(SEQ / 16 / 8, BATCH * HEADS), 256, 0, stream>>>(qh, kh, vt, ctx);

    gemm_wmma<2><<<gg, 256, 0, stream>>>(ctx, woh, bo, d_out, M, N, K);
}
